// ScaledDotProductAttention_91190745629303
// MI455X (gfx1250) — matrix operand order repaired and checked by reading, not compiled, not run on hardware
//
#include <hip/hip_runtime.h>
#include <hip/hip_bf16.h>

// MI455X / gfx1250 flash attention, fp32 I/O.
// Scores: 3xBF16 WMMA emulation (near-fp32). PV: single BF16 WMMA.
// S^T formulation (softmax rows on lanes). Block-level LDS staging of K/V
// (bf16 hi/lo planes + transposed V), double-buffered / software-pipelined.
// 32 queries per wave: every K/V fragment ds_load and every staged byte is
// reused by two query sub-tiles -> ds and VALU per WMMA halved (XDL-bound).

typedef __attribute__((ext_vector_type(16))) __bf16 v16bf;
typedef __attribute__((ext_vector_type(8)))  float  v8f;

#define LOG2E 1.44269504088896340736f
#define KPAD 72   // K row stride in bf16 (64 dims + 8 pad): conflict-free b128 reads
#define VPAD 40   // V^T row stride in bf16 (32 keys + 8 pad)

union BF8x { int4 i4; __bf16 b[8]; };
union Frag { int4 i4[2]; v16bf v; };

__device__ __forceinline__ v8f wmma_bf16(v16bf a, v16bf b, v8f c) {
  return __builtin_amdgcn_wmma_f32_16x16x32_bf16(
      false, a, false, b, (short)0, c, false, false);
}

__device__ __forceinline__ void split_bf16(float x, __bf16& hi, __bf16& lo) {
  hi = (__bf16)x;
  lo = (__bf16)(x - (float)hi);
}

__device__ __forceinline__ float bcast(float x, int byteIdx) {
  return __int_as_float(__builtin_amdgcn_ds_bpermute(byteIdx, __float_as_int(x)));
}

// Geometry: B*H=64, S=2048, D=64. Wave = 32-query tile (2 sub-tiles of 16);
// block = 8 waves = 256 consecutive queries of one (b,h); 512 blocks.
__global__ __launch_bounds__(256)
void ScaledDotProductAttention_91190745629303_kernel(
    const float* __restrict__ q, const float* __restrict__ k,
    const float* __restrict__ v, float* __restrict__ out) {
  __shared__ alignas(16) __bf16 sKhi[2][32][KPAD];
  __shared__ alignas(16) __bf16 sKlo[2][32][KPAD];
  __shared__ alignas(16) __bf16 sVt [2][64][VPAD];

  const int t    = threadIdx.x;
  const int lane = t & 31;
  const int wave = t >> 5;
  const int l16  = lane & 15;
  const int hf   = lane >> 4;              // half-wave (0/1)
  const int bidx_lo = l16 << 2;            // ds_bpermute byte index: lane l16
  const int bidx_hi = (16 + l16) << 2;     // ds_bpermute byte index: lane 16+l16

  const int tile = blockIdx.x * 8 + wave;  // global 32-query tile
  const int bh   = tile >> 6;              // 64 tiles per (b,h)
  const int qt   = tile & 63;
  const long long base = (long long)bh * (2048LL * 64LL);
  const float* qp = q + base;
  const float* kp = k + base;
  const float* vp = v + base;
  float*       op = out + base;

  // ---- staging work assignment (whole block cooperates) ----
  const int skey = t >> 3;                 // K: key 0..31
  const int sdim = (t & 7) * 8;            // K: dim base
  const int vdim = t & 63;                 // V: dim
  const int vkg  = (t >> 6) * 8;           // V: key group base

  // ---- Q^T as WMMA B fragments (dims x queries), hi/lo bf16 split ----
  int qrow[2];
  qrow[0] = qt * 32 + l16;
  qrow[1] = qt * 32 + 16 + l16;
  v16bf qhi[2][2], qlo[2][2];              // [sub][dim-group]
  #pragma unroll
  for (int s = 0; s < 2; ++s) {
    #pragma unroll
    for (int g = 0; g < 2; ++g) {
      const float* src = qp + (long long)qrow[s] * 64 + 32 * g + 16 * hf;
      #pragma unroll
      for (int c4 = 0; c4 < 4; ++c4) {
        float4 f = ((const float4*)src)[c4];
        float fs[4] = {f.x, f.y, f.z, f.w};
        #pragma unroll
        for (int j = 0; j < 4; ++j) {
          __bf16 hi, lo; split_bf16(fs[j], hi, lo);
          qhi[s][g][c4 * 4 + j] = hi;
          qlo[s][g][c4 * 4 + j] = lo;
        }
      }
    }
  }

  // ---- online softmax state (per lane = per query, per sub-tile) ----
  float m[2]  = {-__builtin_inff(), -__builtin_inff()};
  float mb[2] = {-__builtin_inff(), -__builtin_inff()};  // m * LOG2E
  float lsum[2] = {0.f, 0.f};
  v8f acc[2][4];              // O^T: (dim = 16f + r + 8*hf, query = l16)
  #pragma unroll
  for (int s = 0; s < 2; ++s)
    #pragma unroll
    for (int f = 0; f < 4; ++f)
      #pragma unroll
      for (int r = 0; r < 8; ++r) acc[s][f][r] = 0.f;

  // ---- prologue: stage tile 0 into buffer 0 ----
  {
    const float* ks = kp + (long long)skey * 64 + sdim;
    float4 f0 = ((const float4*)ks)[0];
    float4 f1 = ((const float4*)ks)[1];
    float kr[8] = {f0.x,f0.y,f0.z,f0.w,f1.x,f1.y,f1.z,f1.w};
    BF8x hb, lb;
    #pragma unroll
    for (int j = 0; j < 8; ++j) split_bf16(kr[j], hb.b[j], lb.b[j]);
    *(int4*)&sKhi[0][skey][sdim] = hb.i4;
    *(int4*)&sKlo[0][skey][sdim] = lb.i4;
    BF8x vb;
    #pragma unroll
    for (int j = 0; j < 8; ++j)
      vb.b[j] = (__bf16)vp[(long long)(vkg + j) * 64 + vdim];
    *(int4*)&sVt[0][vdim][vkg] = vb.i4;
  }
  __syncthreads();

  for (int kt = 0; kt < 64; ++kt) {        // 64 tiles x 32 keys
    const int cur = kt & 1;
    const int nxt = cur ^ 1;
    const bool more = (kt + 1) < 64;

    // -- issue next tile's global loads early (hide under WMMA) --
    float kreg[8], vreg[8];
    if (more) {
      const int kb2 = (kt + 1) * 32;
      const float* ks = kp + (long long)(kb2 + skey) * 64 + sdim;
      float4 f0 = ((const float4*)ks)[0];
      float4 f1 = ((const float4*)ks)[1];
      kreg[0]=f0.x; kreg[1]=f0.y; kreg[2]=f0.z; kreg[3]=f0.w;
      kreg[4]=f1.x; kreg[5]=f1.y; kreg[6]=f1.z; kreg[7]=f1.w;
      #pragma unroll
      for (int j = 0; j < 8; ++j)
        vreg[j] = vp[(long long)(kb2 + vkg + j) * 64 + vdim];
    }

    // ---- S^T = K * Q^T: two 16-key fragments x two query sub-tiles ----
    v8f st[2][2];                          // [sub][key-fragment]
    #pragma unroll
    for (int h = 0; h < 2; ++h) {
      v8f c0 = {0.f,0.f,0.f,0.f,0.f,0.f,0.f,0.f};
      v8f c1 = {0.f,0.f,0.f,0.f,0.f,0.f,0.f,0.f};
      #pragma unroll
      for (int g = 0; g < 2; ++g) {
        const int key = 16 * h + l16;
        const int db  = 32 * g + 16 * hf;
        Frag fhi, flo;
        fhi.i4[0] = *(const int4*)&sKhi[cur][key][db];
        fhi.i4[1] = *(const int4*)&sKhi[cur][key][db + 8];
        flo.i4[0] = *(const int4*)&sKlo[cur][key][db];
        flo.i4[1] = *(const int4*)&sKlo[cur][key][db + 8];
        c0 = wmma_bf16(flo.v, qhi[0][g], c0);
        c0 = wmma_bf16(fhi.v, qlo[0][g], c0);
        c0 = wmma_bf16(fhi.v, qhi[0][g], c0);
        c1 = wmma_bf16(flo.v, qhi[1][g], c1);
        c1 = wmma_bf16(fhi.v, qlo[1][g], c1);
        c1 = wmma_bf16(fhi.v, qhi[1][g], c1);
      }
      st[0][h] = c0;
      st[1][h] = c1;
    }

    // ---- online softmax per sub-tile (rows = lanes) ----
    v16bf pb[2];
    #pragma unroll
    for (int s = 0; s < 2; ++s) {
      float vmax = st[s][0][0];
      #pragma unroll
      for (int r = 0; r < 8; ++r) {
        vmax = fmaxf(vmax, st[s][0][r]);
        vmax = fmaxf(vmax, st[s][1][r]);
      }
      vmax = fmaxf(vmax, __shfl_xor(vmax, 16));

      // Deferred rescale: wave-uniform, taken only when the max grows.
      if (__any(vmax > m[s])) {
        const float m_new = fmaxf(m[s], vmax);
        const float alpha = __builtin_amdgcn_exp2f((m[s] - m_new) * LOG2E);
        lsum[s] *= alpha;
        #pragma unroll
        for (int f = 0; f < 4; ++f)
          #pragma unroll
          for (int r = 0; r < 8; ++r) acc[s][f][r] *= alpha;
        m[s]  = m_new;
        mb[s] = m_new * LOG2E;
      }

      // p = exp2(s*log2e - mb): one FMA + one TRANS per score element.
      float psum = 0.f;
      #pragma unroll
      for (int r = 0; r < 8; ++r) {
        float a0 = __builtin_amdgcn_exp2f(fmaf(st[s][0][r], LOG2E, -mb[s]));
        float a1 = __builtin_amdgcn_exp2f(fmaf(st[s][1][r], LOG2E, -mb[s]));
        st[s][0][r] = a0; st[s][1][r] = a1;
        psum += a0 + a1;
      }
      psum += __shfl_xor(psum, 16);
      lsum[s] += psum;

      // P^T -> B fragment: own lane already holds table-order keys.
      #pragma unroll
      for (int r = 0; r < 8; ++r) {
        pb[s][r]     = (__bf16)st[s][0][r];
        pb[s][r + 8] = (__bf16)st[s][1][r];
      }
    }

    // ---- O^T += V^T * P^T: each V fragment serves both sub-tiles ----
    #pragma unroll
    for (int f = 0; f < 4; ++f) {
      const int dr = 16 * f + l16;
      Frag va;
      va.i4[0] = *(const int4*)&sVt[cur][dr][8 * hf];
      va.i4[1] = *(const int4*)&sVt[cur][dr][16 + 8 * hf];
      acc[0][f] = wmma_bf16(va.v, pb[0], acc[0][f]);
      acc[1][f] = wmma_bf16(va.v, pb[1], acc[1][f]);
    }

    // -- convert & store next tile into the other buffer --
    if (more) {
      BF8x hb, lb;
      #pragma unroll
      for (int j = 0; j < 8; ++j) split_bf16(kreg[j], hb.b[j], lb.b[j]);
      *(int4*)&sKhi[nxt][skey][sdim] = hb.i4;
      *(int4*)&sKlo[nxt][skey][sdim] = lb.i4;
      BF8x vb;
      #pragma unroll
      for (int j = 0; j < 8; ++j) vb.b[j] = (__bf16)vreg[j];
      *(int4*)&sVt[nxt][vdim][vkg] = vb.i4;
    }
    __syncthreads();
  }

  // ---- normalize (1/l) * 64^-0.25 and store O (row-major [q][d]) ----
  #pragma unroll
  for (int s = 0; s < 2; ++s) {
    const float scale = 0.35355339059327373f / lsum[s];
    float* orow = op + (long long)qrow[s] * 64;
    #pragma unroll
    for (int f = 0; f < 4; ++f) {
      float4 o0, o1;
      o0.x = acc[s][f][0] * scale; o0.y = acc[s][f][1] * scale;
      o0.z = acc[s][f][2] * scale; o0.w = acc[s][f][3] * scale;
      o1.x = acc[s][f][4] * scale; o1.y = acc[s][f][5] * scale;
      o1.z = acc[s][f][6] * scale; o1.w = acc[s][f][7] * scale;
      float* dst = orow + 16 * f + 8 * hf;
      ((float4*)dst)[0] = o0;
      ((float4*)dst)[1] = o1;
    }
  }
}

extern "C" void kernel_launch(void* const* d_in, const int* in_sizes, int n_in,
                              void* d_out, int out_size, void* d_ws, size_t ws_size,
                              hipStream_t stream) {
  const float* q = (const float*)d_in[0];
  const float* k = (const float*)d_in[1];
  const float* v = (const float*)d_in[2];
  float* out = (float*)d_out;
  // 4096 32-query tiles / 8 waves per block = 512 blocks of 256 threads.
  dim3 grid(512), block(256);
  ScaledDotProductAttention_91190745629303_kernel<<<grid, block, 0, stream>>>(q, k, v, out);
}
